// GATModel_61933428408488
// MI455X (gfx1250) — matrix operand order repaired and checked by reading, not compiled, not run on hardware
//
#include <hip/hip_runtime.h>
#include <hip/hip_bf16.h>
#include <cfloat>

typedef __attribute__((ext_vector_type(16))) _Float16 v16h;
typedef __attribute__((ext_vector_type(8)))  _Float16 v8h;
typedef __attribute__((ext_vector_type(8)))  float    v8f;

#define D64 64

// ---------------------------------------------------------------- utilities

__global__ void fill_kernel(float* __restrict__ p, float val, int count) {
    int i = blockIdx.x * blockDim.x + threadIdx.x;
    if (i < count) p[i] = val;
}

// atomic float max via int/uint monotonicity trick
__device__ __forceinline__ void atomicMaxF(float* addr, float val) {
    if (!(val < 0.0f)) atomicMax((int*)addr, __float_as_int(val));
    else               atomicMin((unsigned int*)addr, __float_as_uint(val));
}

__device__ __forceinline__ void edge_sd(const int* __restrict__ ei, int E, int e,
                                        int& s, int& d) {
    if (e < E) { s = ei[e]; d = ei[E + e]; }
    else       { s = e - E; d = e - E; }          // self loops appended
}

// -------------------------------------------------- W packing for WMMA B-layout
// B (32x16 f16 per k-step): lane l, vgpr v, half h  <->  K = 2(v&3)+h+16(v>>2)+8*(l/16)+32*ks,
// col = tn*16 + l%16.  Stored so each lane reads 16 consecutive halves (32B).
__global__ void pack_w_kernel(const float* __restrict__ W /*64x64 row-major [K][n]*/,
                              _Float16* __restrict__ Bp /*2048*2 halves*/) {
    int idx = blockIdx.x * blockDim.x + threadIdx.x;   // 0..2047
    if (idx >= 2048) return;
    int v    = idx & 7;
    int lane = (idx >> 3) & 31;
    int blk  = idx >> 8;          // blk = tn*2 + ks, 0..7
    int ks   = blk & 1;
    int tn   = blk >> 1;
    int lg   = lane >> 4;
    int lr   = lane & 15;
    int K    = 2 * (v & 3) + 16 * (v >> 2) + 8 * lg + 32 * ks;
    int n    = tn * 16 + lr;
    Bp[idx * 2 + 0] = (_Float16)W[(K + 0) * D64 + n];
    Bp[idx * 2 + 1] = (_Float16)W[(K + 1) * D64 + n];
}

// -------------------------------------------------- input concat + f16 convert
__global__ void build_xh_kernel(const float* __restrict__ ue, const float* __restrict__ ie,
                                _Float16* __restrict__ xh, int NU, int total /*N*64*/) {
    int idx = blockIdx.x * blockDim.x + threadIdx.x;
    if (idx >= total) return;
    int n = idx >> 6, f = idx & 63;
    float v = (n < NU) ? ue[idx] : ie[(n - NU) * D64 + f];
    xh[idx] = (_Float16)v;
}

// -------------------------------------------------- WMMA GEMM: H = Ah(f16) @ Bp
// one wave per 16x16 output tile; K=64 as two 16x16x32 WMMAs. rows must be %16==0.
__global__ __launch_bounds__(256) void gemm64_wmma(
    const _Float16* __restrict__ Ah,  // [rows x 64] row-major f16
    const _Float16* __restrict__ Bp,  // packed B layout (4096 halves)
    float* __restrict__ Hout,         // [rows x 64] f32
    int row_tiles) {
    int wave = (blockIdx.x * blockDim.x + threadIdx.x) >> 5;
    int lane = threadIdx.x & 31;
    if (wave >= row_tiles * 4) return;          // wave-uniform: EXEC stays all-ones
    int tm = wave >> 2;
    int tn = wave & 3;
    int lg = lane >> 4;          // lane group (0/1)
    int lr = lane & 15;
    const _Float16* abase = Ah + (tm * 16 + lr) * D64 + 8 * lg;
    v8f c = {};
#pragma unroll
    for (int ks = 0; ks < 2; ++ks) {
        // ISA 16-bit A layout: lanes 0-15 hold K in [0,8)∪[16,24), lanes 16-31 +8
        v8h alo = *(const v8h*)(abase + ks * 32);
        v8h ahi = *(const v8h*)(abase + ks * 32 + 16);
        v16h a;
#pragma unroll
        for (int i = 0; i < 8; ++i) { a[i] = alo[i]; a[8 + i] = ahi[i]; }
        v16h b = *(const v16h*)(Bp + ((tn * 2 + ks) * 32 + lane) * 16);
        c = __builtin_amdgcn_wmma_f32_16x16x32_f16(false, a, false, b,
                                                   (short)0, c, false, false);
    }
    // C/D layout: vgpr v, lane l -> row tm*16 + v + 8*(l/16), col tn*16 + l%16
    float* obase = Hout + (tm * 16 + 8 * lg) * D64 + tn * 16 + lr;
#pragma unroll
    for (int v = 0; v < 8; ++v) obase[v * D64] = c[v];
}

// -------------------------------------------------- per-node attention logits
__global__ void alpha_kernel(const float* __restrict__ Hf, const float* __restrict__ a_src,
                             const float* __restrict__ a_dst, float* __restrict__ as,
                             float* __restrict__ ad, int N, int H, int F) {
    int n = blockIdx.x * blockDim.x + threadIdx.x;
    if (n >= N) return;
    for (int hd = 0; hd < H; ++hd) {
        float s = 0.f, d = 0.f;
        for (int f = 0; f < F; ++f) {
            float v = Hf[n * D64 + hd * F + f];
            s += v * a_src[hd * F + f];
            d += v * a_dst[hd * F + f];
        }
        as[n * H + hd] = s;
        ad[n * H + hd] = d;
    }
}

// -------------------------------------------------- edge passes
__global__ void edge_max_kernel(const int* __restrict__ ei, int E, int N, int H,
                                const float* __restrict__ as, const float* __restrict__ ad,
                                float* __restrict__ m) {
    int idx = blockIdx.x * blockDim.x + threadIdx.x;
    if (idx >= (E + N) * H) return;
    int e = idx / H, h = idx % H;
    int s, d; edge_sd(ei, E, e, s, d);
    float v = as[s * H + h] + ad[d * H + h];
    v = (v > 0.f) ? v : 0.2f * v;                  // leaky relu
    atomicMaxF(&m[d * H + h], v);
}

__global__ void edge_exp_kernel(const int* __restrict__ ei, int E, int N, int H,
                                const float* __restrict__ as, const float* __restrict__ ad,
                                const float* __restrict__ m, float* __restrict__ alphaBuf,
                                float* __restrict__ denom) {
    int idx = blockIdx.x * blockDim.x + threadIdx.x;
    if (idx >= (E + N) * H) return;
    int e = idx / H, h = idx % H;
    int s, d; edge_sd(ei, E, e, s, d);
    float v = as[s * H + h] + ad[d * H + h];
    v = (v > 0.f) ? v : 0.2f * v;
    float p = expf(v - m[d * H + h]);
    alphaBuf[idx] = p;
    atomicAdd(&denom[d * H + h], p);
}

__global__ void edge_agg_kernel(const int* __restrict__ ei, int E, int N, int H, int F,
                                const float* __restrict__ alphaBuf,
                                const float* __restrict__ denom,
                                const float* __restrict__ Hf, float* __restrict__ agg) {
    int idx = blockIdx.x * blockDim.x + threadIdx.x;
    if (idx >= (E + N) * D64) return;
    int e = idx >> 6, f = idx & 63;
    int s, d; edge_sd(ei, E, e, s, d);
    int h = f / F;
    float w = alphaBuf[e * H + h] / denom[d * H + h];
    atomicAdd(&agg[d * D64 + f], Hf[s * D64 + f] * w);   // L2-resident (38MB << 192MB)
}

// -------------------------------------------------- bias + ELU (+ optional f16 copy)
__global__ void finalize_kernel(const float* __restrict__ agg, const float* __restrict__ bias,
                                float* __restrict__ outF, _Float16* __restrict__ outH,
                                int total /*N*64*/) {
    int idx = blockIdx.x * blockDim.x + threadIdx.x;
    if (idx >= total) return;
    int f = idx & 63;
    float y = agg[idx] + bias[f];
    y = (y > 0.f) ? y : (expf(y) - 1.f);           // elu
    if (outF) outF[idx] = y;
    if (outH) outH[idx] = (_Float16)y;
}

// -------------------------------------------------- readout
__global__ void predict_kernel(const float* __restrict__ xf, const int* __restrict__ ui,
                               const int* __restrict__ ii, float* __restrict__ out,
                               int B, int NU) {
    int b = blockIdx.x * blockDim.x + threadIdx.x;
    if (b >= B) return;
    const float* u = xf + (size_t)ui[b] * D64;
    const float* v = xf + (size_t)(NU + ii[b]) * D64;
    float s = 0.f;
    for (int f = 0; f < D64; ++f) s += u[f] * v[f];
    out[b] = 1.f / (1.f + expf(-s));
}

// ================================================================ launcher

static inline int cdiv(long long a, int b) { return (int)((a + b - 1) / b); }

extern "C" void kernel_launch(void* const* d_in, const int* in_sizes, int n_in,
                              void* d_out, int out_size, void* d_ws, size_t ws_size,
                              hipStream_t stream) {
    const int*   ei      = (const int*)d_in[0];     // edge_index [2,E]
    const int*   bu      = (const int*)d_in[1];     // batch_user_idx
    const int*   bi      = (const int*)d_in[2];     // batch_item_idx
    const float* ue      = (const float*)d_in[3];   // user_emb
    const float* ie      = (const float*)d_in[4];   // item_emb
    const float* W1      = (const float*)d_in[5];
    const float* as1w    = (const float*)d_in[6];
    const float* ad1w    = (const float*)d_in[7];
    const float* b1      = (const float*)d_in[8];
    const float* W2      = (const float*)d_in[9];
    const float* as2w    = (const float*)d_in[10];
    const float* ad2w    = (const float*)d_in[11];
    const float* b2      = (const float*)d_in[12];
    float* out = (float*)d_out;

    const int E  = in_sizes[0] / 2;
    const int B  = in_sizes[1];
    const int NU = in_sizes[3] / D64;
    const int NI = in_sizes[4] / D64;
    const int N  = NU + NI;                 // 150000 (divisible by 16)
    const int EP = E + N;
    const int H1 = 4, F1v = 16;
    const int row_tiles = N / 16;

    // ---- carve scratch
    char* ws = (char*)d_ws;
    size_t o = 0;
    auto carve = [&](size_t bytes) -> char* {
        char* p = ws + o;
        o = (o + bytes + 255) & ~(size_t)255;
        return p;
    };
    _Float16* xh     = (_Float16*)carve((size_t)N * D64 * 2);
    float*    hbuf   = (float*)   carve((size_t)N * D64 * 4);   // h1 then h2
    float*    as     = (float*)   carve((size_t)N * 4 * 4);
    float*    ad     = (float*)   carve((size_t)N * 4 * 4);
    float*    mbuf   = (float*)   carve((size_t)N * 4 * 4);
    float*    denom  = (float*)   carve((size_t)N * 4 * 4);
    float*    alphaB = (float*)   carve((size_t)EP * 4 * 4);
    float*    agg    = (float*)   carve((size_t)N * D64 * 4);
    _Float16* x2h    = (_Float16*)carve((size_t)N * D64 * 2);
    float*    xfin   = (float*)   carve((size_t)N * D64 * 4);
    _Float16* wp1    = (_Float16*)carve(4096 * 2);
    _Float16* wp2    = (_Float16*)carve(4096 * 2);
    (void)ws_size;

    const int T = 256;
    const int gemm_blocks = cdiv((long long)row_tiles * 4 * 32, T);

    // ---- prep
    pack_w_kernel<<<cdiv(2048, T), T, 0, stream>>>(W1, wp1);
    pack_w_kernel<<<cdiv(2048, T), T, 0, stream>>>(W2, wp2);
    build_xh_kernel<<<cdiv((long long)N * D64, T), T, 0, stream>>>(ue, ie, xh, NU, N * D64);

    // ================= layer 1 (H=4, F=16) =================
    gemm64_wmma<<<gemm_blocks, T, 0, stream>>>(xh, wp1, hbuf, row_tiles);
    alpha_kernel<<<cdiv(N, T), T, 0, stream>>>(hbuf, as1w, ad1w, as, ad, N, H1, F1v);
    fill_kernel<<<cdiv((long long)N * H1, T), T, 0, stream>>>(mbuf, -FLT_MAX, N * H1);
    fill_kernel<<<cdiv((long long)N * H1, T), T, 0, stream>>>(denom, 0.f, N * H1);
    fill_kernel<<<cdiv((long long)N * D64, T), T, 0, stream>>>(agg, 0.f, N * D64);
    edge_max_kernel<<<cdiv((long long)EP * H1, T), T, 0, stream>>>(ei, E, N, H1, as, ad, mbuf);
    edge_exp_kernel<<<cdiv((long long)EP * H1, T), T, 0, stream>>>(ei, E, N, H1, as, ad, mbuf, alphaB, denom);
    edge_agg_kernel<<<cdiv((long long)EP * D64, T), T, 0, stream>>>(ei, E, N, H1, F1v, alphaB, denom, hbuf, agg);
    finalize_kernel<<<cdiv((long long)N * D64, T), T, 0, stream>>>(agg, b1, nullptr, x2h, N * D64);

    // ================= layer 2 (H=1, F=64) =================
    gemm64_wmma<<<gemm_blocks, T, 0, stream>>>(x2h, wp2, hbuf, row_tiles);
    alpha_kernel<<<cdiv(N, T), T, 0, stream>>>(hbuf, as2w, ad2w, as, ad, N, 1, D64);
    fill_kernel<<<cdiv(N, T), T, 0, stream>>>(mbuf, -FLT_MAX, N);
    fill_kernel<<<cdiv(N, T), T, 0, stream>>>(denom, 0.f, N);
    fill_kernel<<<cdiv((long long)N * D64, T), T, 0, stream>>>(agg, 0.f, N * D64);
    edge_max_kernel<<<cdiv(EP, T), T, 0, stream>>>(ei, E, N, 1, as, ad, mbuf);
    edge_exp_kernel<<<cdiv(EP, T), T, 0, stream>>>(ei, E, N, 1, as, ad, mbuf, alphaB, denom);
    edge_agg_kernel<<<cdiv((long long)EP * D64, T), T, 0, stream>>>(ei, E, N, 1, D64, alphaB, denom, hbuf, agg);
    finalize_kernel<<<cdiv((long long)N * D64, T), T, 0, stream>>>(agg, b2, xfin, nullptr, N * D64);

    // ================= readout =================
    predict_kernel<<<cdiv(B, T), T, 0, stream>>>(xfin, bu, bi, out, B, NU);
}
